// GraphConvolutionalTransformer_37941741093631
// MI455X (gfx1250) — compile-verified, matrix operand order checked by reading, not run on hardware
//
#include <hip/hip_runtime.h>

#define BIGF 3.4028234e+38f
#define EPSF 1e-5f

typedef __attribute__((ext_vector_type(16))) __bf16 v16bf;
typedef __attribute__((ext_vector_type(8)))  __bf16 v8bf;
typedef __attribute__((ext_vector_type(8)))  float  v8f;

// ---------------------------------------------------------------------------
// WMMA helpers
// ---------------------------------------------------------------------------
// CDNA5 16-bit A fragment (16x32, MxK): lane L (L<16) holds row M=L, halves:
//   VGPR0..3 = K[h*8 .. h*8+7], VGPR4..7 = K[16+h*8 .. 16+h*8+7], h = L>>4.
// Both runs are contiguous in K -> two 16-byte loads per lane.
// B fragment (32x16, KxN) mirrors this with lane = column n. Since every GEMM
// here is Y[m,n] = sum_k A[m,k]*Bsrc[n,k] (row-major A and Bsrc), both
// fragments are K-contiguous loads: no transpose needed anywhere.
static __device__ __forceinline__ v16bf ld_frag(const __bf16* p) {
  v8bf lo = *(const v8bf*)(p);
  v8bf hi = *(const v8bf*)(p + 16);
  return __builtin_shufflevector(lo, hi, 0, 1, 2, 3, 4, 5, 6, 7,
                                         8, 9, 10, 11, 12, 13, 14, 15);
}

static __device__ __forceinline__ v8f wmma_bf16(v16bf a, v16bf b, v8f c) {
  // (neg_a, A, neg_b, B, c_mod, C, reuse_a, reuse_b)
  return __builtin_amdgcn_wmma_f32_16x16x32_bf16(false, a, false, b,
                                                 (short)0, c, false, false);
}

enum { MODE_BF16 = 0, MODE_VT = 1, MODE_PRE = 2, MODE_F32 = 3,
       MODE_FFN1 = 4, MODE_FFN2 = 5 };

// ---------------------------------------------------------------------------
// Generic WMMA GEMM: Y[m,n] = sum_k A[m,k] * Bsrc[n,k]   (i.e. A @ Bsrc^T)
// Block = 256 threads = 8 waves arranged 2(M) x 4(N); wave tile = 32x32.
// Grid  = (Nc/128, M/64, batches). All dims divide exactly in this problem.
// ---------------------------------------------------------------------------
template<int MODE>
__global__ __launch_bounds__(256)
void gemm_wmma_kernel(const __bf16* __restrict__ A,
                      const __bf16* __restrict__ B,
                      float* __restrict__ Cf,
                      __bf16* __restrict__ Cb,
                      const float* __restrict__ bias,
                      const float* __restrict__ masks,
                      const float* __restrict__ guide,
                      int M, int Nc, int K, int lda, int ldb,
                      long long strideA, long long strideB, long long strideC,
                      float scale)
{
  const int z = blockIdx.z;
  A += (size_t)z * strideA;
  B += (size_t)z * strideB;

  const int lane = threadIdx.x & 31;
  const int w    = threadIdx.x >> 5;
  const int wm = w & 1, wn = w >> 1;
  const int m0 = blockIdx.y * 64 + wm * 32;
  const int n0 = blockIdx.x * 128 + wn * 32;

  const int r16  = lane & 15;
  const int half = lane >> 4;

  const __bf16* a0 = A + (size_t)(m0 + r16) * lda + half * 8;
  const __bf16* a1 = a0 + (size_t)16 * lda;
  const __bf16* b0 = B + (size_t)(n0 + r16) * ldb + half * 8;
  const __bf16* b1 = b0 + (size_t)16 * ldb;

  v8f c00 = {}, c01 = {}, c10 = {}, c11 = {};
  for (int k0 = 0; k0 < K; k0 += 32) {
    __builtin_prefetch(a0 + k0 + 32, 0, 3);   // global_prefetch_b8
    v16bf fa0 = ld_frag(a0 + k0);
    v16bf fa1 = ld_frag(a1 + k0);
    v16bf fb0 = ld_frag(b0 + k0);
    v16bf fb1 = ld_frag(b1 + k0);
    c00 = wmma_bf16(fa0, fb0, c00);
    c01 = wmma_bf16(fa0, fb1, c01);
    c10 = wmma_bf16(fa1, fb0, c10);
    c11 = wmma_bf16(fa1, fb1, c11);
  }

  // C tile layout: lane holds col = n0+tj*16+(lane&15); VGPR r -> row
  // m0+ti*16 + r + 8*(lane>>4).
  const int colb  = lane & 15;
  const int rbase = (lane >> 4) * 8;
  const v8f acc[2][2] = {{c00, c01}, {c10, c11}};

#pragma unroll
  for (int ti = 0; ti < 2; ++ti) {
#pragma unroll
    for (int tj = 0; tj < 2; ++tj) {
      const v8f c   = acc[ti][tj];
      const int colg = n0 + tj * 16 + colb;
      const int rowb = m0 + ti * 16 + rbase;
      if constexpr (MODE == MODE_VT) {
        // store V transposed per 512-row batch: vT[b][col][mloc]
        const int bb = rowb >> 9;
        const int ml = rowb & 511;
        v8bf pk;
#pragma unroll
        for (int r = 0; r < 8; ++r) pk[r] = (__bf16)c[r];
        *(v8bf*)(Cb + (size_t)bb * 131072 + (size_t)colg * 512 + ml) = pk;
      } else {
#pragma unroll
        for (int r = 0; r < 8; ++r) {
          const int rowg = rowb + r;
          float v = c[r];
          if constexpr (MODE == MODE_FFN1 || MODE == MODE_FFN2) v += bias[colg];
          if constexpr (MODE == MODE_FFN1) v = fmaxf(v, 0.0f);
          if constexpr (MODE == MODE_PRE) {
            const float am = (masks[(size_t)z * 512 + colg] != 0.0f) ? 0.0f : BIGF;
            const float im =
                (guide[((size_t)z * 512 + rowg) * 512 + colg] != 0.0f) ? 0.0f : BIGF;
            v = v * scale - am - im;
          }
          if constexpr (MODE == MODE_BF16 || MODE == MODE_FFN1)
            Cb[(size_t)z * strideC + (size_t)rowg * Nc + colg] = (__bf16)v;
          else
            Cf[(size_t)z * strideC + (size_t)rowg * Nc + colg] = v;
        }
      }
    }
  }
}

// ---------------------------------------------------------------------------
// wave32 reductions (ds_swizzle/permute) + LDS cross-wave combine
// ---------------------------------------------------------------------------
static __device__ __forceinline__ float wred_max(float v) {
#pragma unroll
  for (int m = 16; m > 0; m >>= 1) v = fmaxf(v, __shfl_xor(v, m, 32));
  return v;
}
static __device__ __forceinline__ float wred_sum(float v) {
#pragma unroll
  for (int m = 16; m > 0; m >>= 1) v += __shfl_xor(v, m, 32);
  return v;
}

// softmax over 512-wide rows; in-place on f32, also emits bf16 copy for WMMA
__global__ __launch_bounds__(256)
void softmax_kernel(float* __restrict__ att, __bf16* __restrict__ attb)
{
  __shared__ float sred[8];
  const size_t row = blockIdx.x;
  float* p = att + row * 512;
  const int t = threadIdx.x;
  float v0 = p[t], v1 = p[t + 256];

  float m = wred_max(fmaxf(v0, v1));
  if ((t & 31) == 0) sred[t >> 5] = m;
  __syncthreads();
  float bm = sred[0];
#pragma unroll
  for (int i = 1; i < 8; ++i) bm = fmaxf(bm, sred[i]);
  __syncthreads();

  float e0 = __expf(v0 - bm), e1 = __expf(v1 - bm);
  float s = wred_sum(e0 + e1);
  if ((t & 31) == 0) sred[t >> 5] = s;
  __syncthreads();
  float bs = 0.0f;
#pragma unroll
  for (int i = 0; i < 8; ++i) bs += sred[i];
  const float inv = 1.0f / bs;
  const float o0 = e0 * inv, o1 = e1 * inv;
  p[t] = o0;
  p[t + 256] = o1;
  attb[row * 512 + t] = (__bf16)o0;
  attb[row * 512 + t + 256] = (__bf16)o1;
}

// WHICH==1: LN(A + masks*Bx) -> outf + bf16   (post-attention)
// WHICH==2: LN(A + Bx)        -> outf          (post-FFN)
template<int WHICH>
__global__ __launch_bounds__(256)
void layernorm_kernel(const float* __restrict__ A, const float* __restrict__ Bx,
                      const float* __restrict__ masks,
                      float* __restrict__ outf, __bf16* __restrict__ outb)
{
  __shared__ float sred[8];
  const size_t row = blockIdx.x;
  const int h = threadIdx.x;
  const size_t idx = row * 256 + h;
  float val;
  if constexpr (WHICH == 1) val = A[idx] + masks[row] * Bx[idx];
  else                      val = A[idx] + Bx[idx];

  float s = wred_sum(val);
  if ((h & 31) == 0) sred[h >> 5] = s;
  __syncthreads();
  float tot = 0.0f;
#pragma unroll
  for (int i = 0; i < 8; ++i) tot += sred[i];
  const float mu = tot * (1.0f / 256.0f);
  __syncthreads();

  const float d = val - mu;
  float s2 = wred_sum(d * d);
  if ((h & 31) == 0) sred[h >> 5] = s2;
  __syncthreads();
  float tot2 = 0.0f;
#pragma unroll
  for (int i = 0; i < 8; ++i) tot2 += sred[i];
  const float var = tot2 * (1.0f / 256.0f);
  const float o = d * rsqrtf(var + EPSF);
  outf[idx] = o;
  if constexpr (WHICH == 1) outb[idx] = (__bf16)o;
}

// ---------------------------------------------------------------------------
// elementwise kernels
// ---------------------------------------------------------------------------
__global__ void cvt_bf16_kernel(const float* __restrict__ in,
                                __bf16* __restrict__ out, size_t n) {
  size_t i = (size_t)blockIdx.x * blockDim.x + threadIdx.x;
  if (i < n) out[i] = (__bf16)in[i];
}
__global__ void prep_f_kernel(const float* __restrict__ x,
                              const float* __restrict__ masks,
                              __bf16* __restrict__ f) {
  size_t i = (size_t)blockIdx.x * 256 + threadIdx.x;
  f[i] = (__bf16)(masks[i >> 8] * x[i]);
}
__global__ void att0_kernel(const float* __restrict__ prior,
                            float* __restrict__ att, __bf16* __restrict__ attb) {
  size_t i = (size_t)blockIdx.x * 256 + threadIdx.x;
  float v = prior[i];
  att[i] = v;
  attb[i] = (__bf16)v;
}
__global__ void final_kernel(const float* __restrict__ x,
                             const float* __restrict__ masks,
                             float* __restrict__ out) {
  size_t i = (size_t)blockIdx.x * 256 + threadIdx.x;
  out[i] = x[i] * masks[i >> 8];
}

// ---------------------------------------------------------------------------
// host orchestration
// ---------------------------------------------------------------------------
extern "C" void kernel_launch(void* const* d_in, const int* in_sizes, int n_in,
                              void* d_out, int out_size, void* d_ws, size_t ws_size,
                              hipStream_t stream)
{
  const float* features = (const float*)d_in[0];
  const float* masks    = (const float*)d_in[1];
  const float* guide    = (const float*)d_in[2];
  const float* prior    = (const float*)d_in[3];
  const float* WQ = (const float*)d_in[4];
  const float* WK = (const float*)d_in[5];
  const float* WV = (const float*)d_in[6];
  const float* W1 = (const float*)d_in[7];
  const float* b1 = (const float*)d_in[8];
  const float* W2 = (const float*)d_in[9];
  const float* b2 = (const float*)d_in[10];

  constexpr int B = 32, N = 512, H = 256, S = 3;
  constexpr size_t M  = (size_t)B * N;       // 16384 rows
  constexpr size_t FE = M * H;               // feature elements
  constexpr size_t AE = (size_t)B * N * N;   // attention elements per layer

  char* ws = (char*)d_ws;
  size_t off = 0;
  auto carve = [&](size_t bytes) -> char* {
    char* p = ws + off;
    off += (bytes + 255) & ~(size_t)255;
    return p;
  };
  __bf16* f_bf    = (__bf16*)carve(FE * 2);
  __bf16* q_bf    = (__bf16*)carve(FE * 2);   // aliased as ffn1 output
  __bf16* k_bf    = (__bf16*)carve(FE * 2);
  __bf16* vT_bf   = (__bf16*)carve(FE * 2);
  __bf16* att_bf  = (__bf16*)carve(AE * 2);
  float*  postraw = (float*) carve(FE * 4);   // aliased as ffn2 output
  float*  post    = (float*) carve(FE * 4);
  __bf16* post_bf = (__bf16*)carve(FE * 2);
  float*  xbuf    = (float*) carve(FE * 4);
  const size_t WN = (size_t)S * H * H;
  __bf16* WQb = (__bf16*)carve(WN * 2);
  __bf16* WKb = (__bf16*)carve(WN * 2);
  __bf16* WVb = (__bf16*)carve(WN * 2);
  __bf16* W1b = (__bf16*)carve(WN * 2);
  __bf16* W2b = (__bf16*)carve(WN * 2);

  float* out_x   = (float*)d_out;
  float* out_att = (float*)d_out + FE;

  // convert weights fp32 -> bf16 (once per call; deterministic)
  {
    dim3 g((unsigned)((WN + 255) / 256)), b(256);
    cvt_bf16_kernel<<<g, b, 0, stream>>>(WQ, WQb, WN);
    cvt_bf16_kernel<<<g, b, 0, stream>>>(WK, WKb, WN);
    cvt_bf16_kernel<<<g, b, 0, stream>>>(WV, WVb, WN);
    cvt_bf16_kernel<<<g, b, 0, stream>>>(W1, W1b, WN);
    cvt_bf16_kernel<<<g, b, 0, stream>>>(W2, W2b, WN);
  }

  const float scale = 0.0625f;  // 1/sqrt(256)

  for (int i = 0; i < S; ++i) {
    const float* xcur = (i == 0) ? features : xbuf;
    float* atti = out_att + (size_t)i * AE;
    const __bf16* Wq = WQb + (size_t)i * H * H;
    const __bf16* Wk = WKb + (size_t)i * H * H;
    const __bf16* Wv = WVb + (size_t)i * H * H;
    const __bf16* w1 = W1b + (size_t)i * H * H;
    const __bf16* w2 = W2b + (size_t)i * H * H;
    const float* b1i = b1 + (size_t)i * H;
    const float* b2i = b2 + (size_t)i * H;

    // f = masks * x (bf16 operand for all GEMMs)
    prep_f_kernel<<<dim3((unsigned)(FE / 256)), 256, 0, stream>>>(xcur, masks, f_bf);

    // v = f @ WV^T, stored transposed per batch: vT[b][d][n]
    gemm_wmma_kernel<MODE_VT><<<dim3(H / 128, (unsigned)(M / 64), 1), 256, 0, stream>>>(
        f_bf, Wv, nullptr, vT_bf, nullptr, nullptr, nullptr,
        (int)M, H, H, H, H, 0, 0, 0, 0.0f);

    if (i == 0) {
      att0_kernel<<<dim3((unsigned)(AE / 256)), 256, 0, stream>>>(prior, atti, att_bf);
    } else {
      gemm_wmma_kernel<MODE_BF16><<<dim3(H / 128, (unsigned)(M / 64), 1), 256, 0, stream>>>(
          f_bf, Wq, nullptr, q_bf, nullptr, nullptr, nullptr,
          (int)M, H, H, H, H, 0, 0, 0, 0.0f);
      gemm_wmma_kernel<MODE_BF16><<<dim3(H / 128, (unsigned)(M / 64), 1), 256, 0, stream>>>(
          f_bf, Wk, nullptr, k_bf, nullptr, nullptr, nullptr,
          (int)M, H, H, H, H, 0, 0, 0, 0.0f);
      // pre = q @ k^T * scale - attn_mask - inf_mask  (f32, into d_out slot)
      gemm_wmma_kernel<MODE_PRE><<<dim3(N / 128, N / 64, B), 256, 0, stream>>>(
          q_bf, k_bf, atti, nullptr, nullptr, masks, guide,
          N, N, H, H, H, (long long)N * H, (long long)N * H,
          (long long)N * N, scale);
      softmax_kernel<<<dim3(B * N), 256, 0, stream>>>(atti, att_bf);
    }

    // post_raw = att @ v    (A = att[b] 512x512, Bsrc = vT[b] 256x512)
    gemm_wmma_kernel<MODE_F32><<<dim3(H / 128, N / 64, B), 256, 0, stream>>>(
        att_bf, vT_bf, postraw, nullptr, nullptr, nullptr, nullptr,
        N, H, N, N, N, (long long)N * N, (long long)H * N,
        (long long)N * H, 0.0f);

    // post = LN(post_raw + masks*x)
    layernorm_kernel<1><<<dim3((unsigned)M), 256, 0, stream>>>(
        postraw, xcur, masks, post, post_bf);

    // ffn1 = relu(post @ W1^T + b1)   (bf16, reuses q_bf)
    gemm_wmma_kernel<MODE_FFN1><<<dim3(H / 128, (unsigned)(M / 64), 1), 256, 0, stream>>>(
        post_bf, w1, nullptr, q_bf, b1i, nullptr, nullptr,
        (int)M, H, H, H, H, 0, 0, 0, 0.0f);
    // ffn2 = ffn1 @ W2^T + b2         (f32, reuses postraw)
    gemm_wmma_kernel<MODE_FFN2><<<dim3(H / 128, (unsigned)(M / 64), 1), 256, 0, stream>>>(
        q_bf, w2, postraw, nullptr, b2i, nullptr, nullptr,
        (int)M, H, H, H, H, 0, 0, 0, 0.0f);

    // x = LN(ffn2 + post)
    layernorm_kernel<2><<<dim3((unsigned)M), 256, 0, stream>>>(
        postraw, post, nullptr, xbuf, nullptr);
  }

  final_kernel<<<dim3((unsigned)(FE / 256)), 256, 0, stream>>>(xbuf, masks, out_x);
}
